// FFM_2241972928687
// MI455X (gfx1250) — matrix operand order repaired and checked by reading, not compiled, not run on hardware
//
#include <hip/hip_runtime.h>
#include <hip/hip_bf16.h>

// ---------------------------------------------------------------------------
// FFM forward for MI455X (gfx1250, wave32, WMMA bf16 16x16x32)
//   T=4096, IN=512, TRACE=64, CTX=64, OUT=512
// ---------------------------------------------------------------------------

#define T_DIM   4096
#define IN_DIM  512
#define TRACE   64
#define CTX     64
#define OUT_DIM 512
#define MIXK    (2 * TRACE * CTX)   // 8192
#define NPAIR   (TRACE * CTX)       // 4096
#define NCHUNK  32
#define CLEN    (T_DIM / NCHUNK)    // 128
#define LN_EPS  1e-6f

typedef __attribute__((ext_vector_type(16))) __bf16        v16bf;
typedef __attribute__((ext_vector_type(8)))  float         v8f;
typedef __attribute__((ext_vector_type(4)))  unsigned int  u32x4;

union Frag16 { v16bf v; u32x4 u[2]; };

// ---------------------------------------------------------------------------
// Prep: fp32 -> bf16 conversion
// ---------------------------------------------------------------------------
__global__ void k_cvt_bf16(const float* __restrict__ src, __bf16* __restrict__ dst, int n) {
    int i = blockIdx.x * blockDim.x + threadIdx.x;
    if (i < n) dst[i] = (__bf16)src[i];
}

// ---------------------------------------------------------------------------
// Tiled transpose + downconvert: W [K,N] fp32 row-major -> WT [N,K] bf16.
// 32x32 tiles through LDS (padded to kill bank conflicts); both the global
// read and the global write are fully coalesced.  Grid: (K/32, N/32).
// ---------------------------------------------------------------------------
__global__ void k_transpose_w(const float* __restrict__ W, __bf16* __restrict__ WT,
                              int K, int N) {
    __shared__ __bf16 tile[32][33];
    const int kt = blockIdx.x << 5;
    const int nt = blockIdx.y << 5;
    const int tx = threadIdx.x & 31;          // fast dim
    const int ty = threadIdx.x >> 5;          // 0..7
    #pragma unroll
    for (int i = ty; i < 32; i += 8)
        tile[i][tx] = (__bf16)W[(size_t)(kt + i) * N + nt + tx];
    __syncthreads();
    #pragma unroll
    for (int i = ty; i < 32; i += 8)
        WT[(size_t)(nt + i) * K + kt + tx] = tile[tx][i];
}

// ---------------------------------------------------------------------------
// bf16 WMMA GEMM:  C[M,N] = A[M,K] * B[K,N] + bias[N]
//   A  : [M,K] row-major bf16
//   BT : [N,K] row-major bf16 (B transposed)
// Block = 128 threads = 4 waves; each wave computes a 32(M) x 32(N) tile with
// 4 f32 accumulators (4 WMMAs per 8 fragment loads per k-step).
// Fragment layouts per CDNA5 ISA 7.12.2:
//   A 16x32 bf16 : lane l holds row m+(l&15), K = k0 + (l>>4)*8 + {0..7,16..23}
//   B 32x16 bf16 : lane l holds col n+(l&15), K = k0 + (l>>4)*8 + {0..7,16..23}
//   C 16x16 f32  : vgpr r -> row m + r + (l>>4)*8, col n + (l&15)
// ---------------------------------------------------------------------------
__global__ void k_gemm_bf16(const __bf16* __restrict__ A,
                            const __bf16* __restrict__ BT,
                            const float* __restrict__ bias,
                            float* __restrict__ C,
                            int M, int N, int K) {
    const int lane = threadIdx.x & 31;
    const int wave = threadIdx.x >> 5;
    const int ntn  = N >> 5;                       // N / 32 tiles
    const int tile = blockIdx.x * 4 + wave;
    if (tile >= (M >> 5) * ntn) return;
    const int m0 = (tile / ntn) << 5;
    const int n0 = (tile % ntn) << 5;

    const int half = lane >> 4;                    // 0 | 1
    const int l16  = lane & 15;

    const __bf16* __restrict__ Arow0 = A  + (size_t)(m0 + l16) * K;
    const __bf16* __restrict__ Arow1 = A  + (size_t)(m0 + 16 + l16) * K;
    const __bf16* __restrict__ Brow0 = BT + (size_t)(n0 + l16) * K;
    const __bf16* __restrict__ Brow1 = BT + (size_t)(n0 + 16 + l16) * K;
    const int akb = half << 3;                     // A k-sub-offset: 0 | 8
    const int bkb = half << 3;                     // B k-sub-offset: 0 | 8

    v8f acc00 = {}, acc01 = {}, acc10 = {}, acc11 = {};

    for (int k0 = 0; k0 < K; k0 += 32) {
        Frag16 a0, a1, b0, b1;
        a0.u[0] = *(const u32x4*)(Arow0 + k0 + akb);
        a0.u[1] = *(const u32x4*)(Arow0 + k0 + 16 + akb);
        a1.u[0] = *(const u32x4*)(Arow1 + k0 + akb);
        a1.u[1] = *(const u32x4*)(Arow1 + k0 + 16 + akb);
        b0.u[0] = *(const u32x4*)(Brow0 + k0 + bkb);
        b0.u[1] = *(const u32x4*)(Brow0 + k0 + 16 + bkb);
        b1.u[0] = *(const u32x4*)(Brow1 + k0 + bkb);
        b1.u[1] = *(const u32x4*)(Brow1 + k0 + 16 + bkb);
        if (k0 + 32 < K) {                          // near-cache stream prefetch
            __builtin_prefetch(Arow0 + k0 + 32 + akb, 0, 3);
            __builtin_prefetch(Arow1 + k0 + 32 + akb, 0, 3);
            __builtin_prefetch(Brow0 + k0 + 32 + bkb, 0, 3);
            __builtin_prefetch(Brow1 + k0 + 32 + bkb, 0, 3);
        }
        acc00 = __builtin_amdgcn_wmma_f32_16x16x32_bf16(false, a0.v, false, b0.v,
                                                        (short)0, acc00, false, false);
        acc01 = __builtin_amdgcn_wmma_f32_16x16x32_bf16(false, a0.v, false, b1.v,
                                                        (short)0, acc01, false, false);
        acc10 = __builtin_amdgcn_wmma_f32_16x16x32_bf16(false, a1.v, false, b0.v,
                                                        (short)0, acc10, false, false);
        acc11 = __builtin_amdgcn_wmma_f32_16x16x32_bf16(false, a1.v, false, b1.v,
                                                        (short)0, acc11, false, false);
    }

    const int colA = n0 + l16;
    const int colB = n0 + 16 + l16;
    const float bA = bias[colA];
    const float bB = bias[colB];
    #pragma unroll
    for (int r = 0; r < 8; ++r) {
        const int row0 = m0 + r + (half << 3);
        const int row1 = row0 + 16;
        C[(size_t)row0 * N + colA] = acc00[r] + bA;
        C[(size_t)row0 * N + colB] = acc01[r] + bB;
        C[(size_t)row1 * N + colA] = acc10[r] + bA;
        C[(size_t)row1 * N + colB] = acc11[r] + bB;
    }
}

// ---------------------------------------------------------------------------
// gated = (x@W_pre + b_pre) * sigmoid(x@W_gin + b_gin)   (biases already fused)
// ---------------------------------------------------------------------------
__global__ void k_make_gated(const float* __restrict__ pre_o,
                             const float* __restrict__ gin_o,
                             float* __restrict__ gated, int n) {
    int i = blockIdx.x * blockDim.x + threadIdx.x;
    if (i >= n) return;
    gated[i] = pre_o[i] * (1.f / (1.f + __expf(-gin_o[i])));
}

// ---------------------------------------------------------------------------
// Scan phase 1: per-chunk end state (state starts at 0 within a chunk)
//   s_t = g * s_{t-1} + gated[t, tr]     g = exp(-|a_tr|) * (cos b_cx + i sin b_cx)
// ---------------------------------------------------------------------------
__global__ void k_scan_ends(const float* __restrict__ gated,
                            const float* __restrict__ ffa_a,
                            const float* __restrict__ ffa_b,
                            float* __restrict__ ends /*[NCHUNK][NPAIR][2]*/) {
    int idx = blockIdx.x * blockDim.x + threadIdx.x;
    if (idx >= NCHUNK * NPAIR) return;
    const int chunk = idx >> 12;
    const int pair  = idx & (NPAIR - 1);
    const int tr = pair >> 6, cx = pair & 63;
    const float er = __expf(-fabsf(ffa_a[tr]));
    const float b  = ffa_b[cx];
    const float gr = er * __cosf(b), gi = er * __sinf(b);
    float sr = 0.f, si = 0.f;
    const int t0 = chunk * CLEN;
    for (int i = 0; i < CLEN; ++i) {
        const float z  = gated[(size_t)(t0 + i) * TRACE + tr];
        const float nr = sr * gr - si * gi + z;
        const float ni = sr * gi + si * gr;
        sr = nr; si = ni;
    }
    ends[(size_t)idx * 2 + 0] = sr;
    ends[(size_t)idx * 2 + 1] = si;
}

// ---------------------------------------------------------------------------
// Scan phase 2: sequential combine across the 32 chunks (per pair).
//   carry[0] = state0;  carry[c+1] = carry[c] * g^CLEN + ends[c]
// g^128 via 7 complex squarings.
// ---------------------------------------------------------------------------
__global__ void k_scan_carries(const float* __restrict__ ends,
                               const float* __restrict__ state0,
                               const float* __restrict__ ffa_a,
                               const float* __restrict__ ffa_b,
                               float* __restrict__ carries /*[NCHUNK][NPAIR][2]*/) {
    int pair = blockIdx.x * blockDim.x + threadIdx.x;
    if (pair >= NPAIR) return;
    const int tr = pair >> 6, cx = pair & 63;
    const float er = __expf(-fabsf(ffa_a[tr]));
    const float b  = ffa_b[cx];
    float glr = er * __cosf(b), gli = er * __sinf(b);
    #pragma unroll
    for (int i = 0; i < 7; ++i) {                  // g -> g^128
        const float nr = glr * glr - gli * gli;
        const float ni = 2.f * glr * gli;
        glr = nr; gli = ni;
    }
    float cr = state0[(size_t)pair * 2 + 0];
    float ci = state0[(size_t)pair * 2 + 1];
    for (int ch = 0; ch < NCHUNK; ++ch) {
        const size_t o = ((size_t)ch * NPAIR + pair) * 2;
        carries[o + 0] = cr;
        carries[o + 1] = ci;
        const float nr = cr * glr - ci * gli + ends[o + 0];
        const float ni = cr * gli + ci * glr + ends[o + 1];
        cr = nr; ci = ni;
    }
}

// ---------------------------------------------------------------------------
// Scan phase 3: re-run each chunk from its carry; emit z_in as bf16 in the
// reference layout: z_in[t, tr*128 + cx] = Re, z_in[t, tr*128 + 64 + cx] = Im
// (lane index carries cx, so wave stores are 64B-contiguous)
// ---------------------------------------------------------------------------
__global__ void k_scan_write(const float* __restrict__ gated,
                             const float* __restrict__ ffa_a,
                             const float* __restrict__ ffa_b,
                             const float* __restrict__ carries,
                             __bf16* __restrict__ z_in /*[T, 8192]*/) {
    int idx = blockIdx.x * blockDim.x + threadIdx.x;
    if (idx >= NCHUNK * NPAIR) return;
    const int chunk = idx >> 12;
    const int pair  = idx & (NPAIR - 1);
    const int tr = pair >> 6, cx = pair & 63;
    const float er = __expf(-fabsf(ffa_a[tr]));
    const float b  = ffa_b[cx];
    const float gr = er * __cosf(b), gi = er * __sinf(b);
    float sr = carries[(size_t)idx * 2 + 0];
    float si = carries[(size_t)idx * 2 + 1];
    const int t0 = chunk * CLEN;
    for (int i = 0; i < CLEN; ++i) {
        const int t = t0 + i;
        const float z  = gated[(size_t)t * TRACE + tr];
        const float nr = sr * gr - si * gi + z;
        const float ni = sr * gi + si * gr;
        sr = nr; si = ni;
        __bf16* row = z_in + (size_t)t * MIXK + tr * (2 * CTX);
        row[cx]       = (__bf16)sr;
        row[CTX + cx] = (__bf16)si;
    }
}

// ---------------------------------------------------------------------------
// Epilogue: v = zm * sigmoid(gout_raw); LN(v) + skip * (1 - sigmoid(gout_raw))
// One block (256 threads, 2 cols each) per row of 512.
// ---------------------------------------------------------------------------
__global__ void k_epilogue(const float* __restrict__ zm,
                           const float* __restrict__ gout_raw,
                           const float* __restrict__ skipv,
                           float* __restrict__ out) {
    const int row = blockIdx.x;
    const int tid = threadIdx.x;
    __shared__ float sdata[256];
    const size_t base = (size_t)row * OUT_DIM;
    const int c0 = tid, c1 = tid + 256;

    const float go0 = 1.f / (1.f + __expf(-gout_raw[base + c0]));
    const float go1 = 1.f / (1.f + __expf(-gout_raw[base + c1]));
    const float v0 = zm[base + c0] * go0;
    const float v1 = zm[base + c1] * go1;

    sdata[tid] = v0 + v1;
    __syncthreads();
    for (int off = 128; off > 0; off >>= 1) {
        if (tid < off) sdata[tid] += sdata[tid + off];
        __syncthreads();
    }
    const float mu = sdata[0] * (1.f / (float)OUT_DIM);
    __syncthreads();

    const float d0 = v0 - mu, d1 = v1 - mu;
    sdata[tid] = d0 * d0 + d1 * d1;
    __syncthreads();
    for (int off = 128; off > 0; off >>= 1) {
        if (tid < off) sdata[tid] += sdata[tid + off];
        __syncthreads();
    }
    const float rs = rsqrtf(sdata[0] * (1.f / (float)OUT_DIM) + LN_EPS);

    out[base + c0] = d0 * rs + skipv[base + c0] * (1.f - go0);
    out[base + c1] = d1 * rs + skipv[base + c1] * (1.f - go1);
}

// ---------------------------------------------------------------------------
// Host-side orchestration
// ---------------------------------------------------------------------------
static inline void launch_gemm(const __bf16* A, const __bf16* BT, const float* bias,
                               float* C, int M, int N, int K, hipStream_t s) {
    const int waves  = (M >> 5) * (N >> 5);        // 32x32 per wave
    const int blocks = (waves + 3) >> 2;
    k_gemm_bf16<<<blocks, 128, 0, s>>>(A, BT, bias, C, M, N, K);
}

static inline void launch_transpose(const float* W, __bf16* WT, int K, int N,
                                    hipStream_t s) {
    dim3 grid(K >> 5, N >> 5);
    k_transpose_w<<<grid, 256, 0, s>>>(W, WT, K, N);
}

extern "C" void kernel_launch(void* const* d_in, const int* in_sizes, int n_in,
                              void* d_out, int out_size, void* d_ws, size_t ws_size,
                              hipStream_t stream) {
    const float* x      = (const float*)d_in[0];
    const float* state0 = (const float*)d_in[1];
    const float* ffa_a  = (const float*)d_in[2];
    const float* ffa_b  = (const float*)d_in[3];
    const float* W_pre  = (const float*)d_in[4];
    const float* b_pre  = (const float*)d_in[5];
    const float* W_gin  = (const float*)d_in[6];
    const float* b_gin  = (const float*)d_in[7];
    const float* W_gout = (const float*)d_in[8];
    const float* b_gout = (const float*)d_in[9];
    const float* W_skip = (const float*)d_in[10];
    const float* b_skip = (const float*)d_in[11];
    const float* W_mix  = (const float*)d_in[12];
    const float* b_mix  = (const float*)d_in[13];
    float* out = (float*)d_out;

    // ---- workspace carving (256B aligned) ----
    char* p = (char*)d_ws;
    auto alloc = [&](size_t bytes) -> char* {
        char* r = p;
        p += (bytes + 255) & ~(size_t)255;
        return r;
    };
    __bf16* x_bf    = (__bf16*)alloc((size_t)T_DIM * IN_DIM * 2);
    __bf16* WpreT   = (__bf16*)alloc((size_t)TRACE * IN_DIM * 2);
    __bf16* WginT   = (__bf16*)alloc((size_t)TRACE * IN_DIM * 2);
    __bf16* WgoutT  = (__bf16*)alloc((size_t)OUT_DIM * IN_DIM * 2);
    __bf16* WskipT  = (__bf16*)alloc((size_t)OUT_DIM * IN_DIM * 2);
    __bf16* WmixT   = (__bf16*)alloc((size_t)OUT_DIM * MIXK * 2);
    float*  pre_o   = (float*)alloc((size_t)T_DIM * TRACE * 4);
    float*  gin_o   = (float*)alloc((size_t)T_DIM * TRACE * 4);
    float*  gout_o  = (float*)alloc((size_t)T_DIM * OUT_DIM * 4);
    float*  skip_o  = (float*)alloc((size_t)T_DIM * OUT_DIM * 4);
    float*  gated   = (float*)alloc((size_t)T_DIM * TRACE * 4);
    float*  ends    = (float*)alloc((size_t)NCHUNK * NPAIR * 2 * 4);
    float*  carries = (float*)alloc((size_t)NCHUNK * NPAIR * 2 * 4);
    __bf16* z_in    = (__bf16*)alloc((size_t)T_DIM * MIXK * 2);
    float*  zm      = (float*)alloc((size_t)T_DIM * OUT_DIM * 4);
    (void)ws_size; (void)in_sizes; (void)n_in; (void)out_size;

    // ---- prep: bf16 conversion + coalesced LDS-tiled weight transposes ----
    {
        int n = T_DIM * IN_DIM;
        k_cvt_bf16<<<(n + 255) / 256, 256, 0, stream>>>(x, x_bf, n);
    }
    launch_transpose(W_pre,  WpreT,  IN_DIM, TRACE,   stream);
    launch_transpose(W_gin,  WginT,  IN_DIM, TRACE,   stream);
    launch_transpose(W_gout, WgoutT, IN_DIM, OUT_DIM, stream);
    launch_transpose(W_skip, WskipT, IN_DIM, OUT_DIM, stream);
    launch_transpose(W_mix,  WmixT,  MIXK,   OUT_DIM, stream);

    // ---- projection GEMMs (bias fused) ----
    launch_gemm(x_bf, WpreT,  b_pre,  pre_o,  T_DIM, TRACE,   IN_DIM, stream);
    launch_gemm(x_bf, WginT,  b_gin,  gin_o,  T_DIM, TRACE,   IN_DIM, stream);
    launch_gemm(x_bf, WgoutT, b_gout, gout_o, T_DIM, OUT_DIM, IN_DIM, stream);
    launch_gemm(x_bf, WskipT, b_skip, skip_o, T_DIM, OUT_DIM, IN_DIM, stream);

    // ---- gated input ----
    {
        int n = T_DIM * TRACE;
        k_make_gated<<<(n + 255) / 256, 256, 0, stream>>>(pre_o, gin_o, gated, n);
    }

    // ---- 3-phase chunked complex scan ----
    {
        int n = NCHUNK * NPAIR;
        k_scan_ends<<<(n + 255) / 256, 256, 0, stream>>>(gated, ffa_a, ffa_b, ends);
        k_scan_carries<<<(NPAIR + 255) / 256, 256, 0, stream>>>(ends, state0, ffa_a, ffa_b, carries);
        k_scan_write<<<(n + 255) / 256, 256, 0, stream>>>(gated, ffa_a, ffa_b, carries, z_in);
    }

    // ---- dominant mix GEMM: [4096,8192] x [8192,512] ----
    launch_gemm(z_in, WmixT, b_mix, zm, T_DIM, OUT_DIM, MIXK, stream);

    // ---- gated LayerNorm epilogue ----
    k_epilogue<<<T_DIM, 256, 0, stream>>>(zm, gout_o, skip_o, out);
}
